// GCNIIMessageLayer_37022618092149
// MI455X (gfx1250) — matrix operand order repaired and checked by reading, not compiled, not run on hardware
//
#include <hip/hip_runtime.h>
#include <hip/hip_bf16.h>

typedef __attribute__((ext_vector_type(16))) _Float16 v16h;
typedef __attribute__((ext_vector_type(8)))  float    v8f;

#define GD 128  // feature dimension D

static constexpr float kAlpha = 0.1f;
static constexpr float kBeta  = 0.40546510810816438198f;  // log(0.5/1 + 1)

// ---------------------------------------------------------------- degree ----
__global__ void k_deg_init(float* __restrict__ deg, int n) {
  int i = blockIdx.x * blockDim.x + threadIdx.x;
  if (i < n) deg[i] = 1.0f;  // self-loop contribution
}

__global__ void k_deg_count(const long long* __restrict__ dst,
                            float* __restrict__ deg, int e) {
  int i = blockIdx.x * blockDim.x + threadIdx.x;
  if (i < e) atomicAdd(&deg[(int)dst[i]], 1.0f);
}

__global__ void k_dinv(const float* __restrict__ deg,
                       float* __restrict__ dinv, int n) {
  int i = blockIdx.x * blockDim.x + threadIdx.x;
  if (i < n) dinv[i] = rsqrtf(deg[i]);  // deg >= 1 always (self-loops)
}

// -------------------------------------------- agg init (self-loop folded) ---
__global__ void k_selfloop(const float* __restrict__ x,
                           const float* __restrict__ dinv,
                           float* __restrict__ agg, long total) {
  long i = (long)blockIdx.x * blockDim.x + threadIdx.x;
  if (i < total) {
    int row = (int)(i >> 7);
    float d = dinv[row];
    agg[i] = x[i] * (d * d);
  }
}

// --------------------------------------------------- edge scatter (L2-hot) --
// one wave32 per edge; float4 gather per lane; 4 no-return f32 atomics/lane.
// x (51 MB) and agg (51 MB) both fit in the 192 MB L2 -> L2-atomic bound.
__global__ void k_scatter(const float* __restrict__ x,
                          const float* __restrict__ dinv,
                          const long long* __restrict__ src,
                          const long long* __restrict__ dst,
                          float* __restrict__ agg, int e) {
  int gid  = blockIdx.x * blockDim.x + threadIdx.x;
  int wid  = gid >> 5;
  int lane = threadIdx.x & 31;
  if (wid >= e) return;
  int s = (int)src[wid];
  int t = (int)dst[wid];
  float norm = dinv[s] * dinv[t];
  const float4 v = *(const float4*)(x + (size_t)s * GD + lane * 4);
  float* p = agg + (size_t)t * GD + lane * 4;
  atomicAdd(p + 0, v.x * norm);
  atomicAdd(p + 1, v.y * norm);
  atomicAdd(p + 2, v.z * norm);
  atomicAdd(p + 3, v.w * norm);
}

// ------------------------- fused epilogue + WMMA GEMM -----------------------
// out = (1-beta)*Z + Z @ (beta*W), Z = (1-alpha)*agg + alpha*x0
// block = 256 threads (8 waves), each wave owns 16 rows; W^T (f16, pre-scaled
// by beta) and the Z tile staged in LDS. D fragments are written back into
// the Z tile so the global store is 16 coalesced b128 stores per lane.
__global__ __launch_bounds__(256) void k_out_wmma(
    const float* __restrict__ agg, const float* __restrict__ x0,
    const float* __restrict__ W, float* __restrict__ out, int n) {
  __shared__ __align__(32) _Float16 wlds[GD * GD];  // wlds[c*GD + k] = beta*W[k][c]
  __shared__ __align__(32) float    zlds[128 * GD]; // Z tile -> result tile

  const int tid  = threadIdx.x;
  const int wave = tid >> 5;
  const int lane = tid & 31;

  // stage beta * W^T as f16 (contiguous LDS writes; W is tiny & L2-resident)
  for (int j = tid; j < GD * GD; j += 256) {
    int c = j >> 7;
    int k = j & (GD - 1);
    wlds[j] = (_Float16)(kBeta * W[k * GD + c]);
  }

  const int r0     = blockIdx.x * 128 + wave * 16;
  const bool active = (r0 < n);
  const bool full   = (r0 + 16 <= n);  // wave-uniform; true for all active waves
                                       // when n % 16 == 0 (N = 100000)
  const int c4 = lane * 4;
  float* zw = zlds + wave * 16 * GD;   // this wave's private 16x128 region

  if (active) {
    if (full) {
      for (int m = 0; m < 16; ++m) {
        const size_t g = (size_t)(r0 + m) * GD + c4;
        float4 a = *(const float4*)(agg + g);
        float4 b = *(const float4*)(x0 + g);
        float4 z;
        z.x = (1.0f - kAlpha) * a.x + kAlpha * b.x;
        z.y = (1.0f - kAlpha) * a.y + kAlpha * b.y;
        z.z = (1.0f - kAlpha) * a.z + kAlpha * b.z;
        z.w = (1.0f - kAlpha) * a.w + kAlpha * b.w;
        *(float4*)(zw + m * GD + c4) = z;
      }
    } else {
      for (int m = 0; m < 16; ++m) {
        float4 z = make_float4(0.f, 0.f, 0.f, 0.f);
        if (r0 + m < n) {  // wave-uniform compare
          const size_t g = (size_t)(r0 + m) * GD + c4;
          float4 a = *(const float4*)(agg + g);
          float4 b = *(const float4*)(x0 + g);
          z.x = (1.0f - kAlpha) * a.x + kAlpha * b.x;
          z.y = (1.0f - kAlpha) * a.y + kAlpha * b.y;
          z.z = (1.0f - kAlpha) * a.z + kAlpha * b.z;
          z.w = (1.0f - kAlpha) * a.w + kAlpha * b.w;
        }
        *(float4*)(zw + m * GD + c4) = z;
      }
    }
  }
  __syncthreads();
  if (!active) return;

  const int mrow = lane & 15;
  const int hi   = lane >> 4;  // 0: lanes 0-15, 1: lanes 16-31

  // A fragments (16x32 f16), contiguous K layout (matches B):
  //   lane<16: K = 32k+{0..15} in a[0..15]
  //   lane>=16: K = 32k+{16..31}
  const float* zr = zw + mrow * GD;
  const int off = hi * 16;
  v16h afrag[4];
#pragma unroll
  for (int k = 0; k < 4; ++k) {
    v16h a;
#pragma unroll
    for (int j = 0; j < 8; ++j) {
      a[j]     = (_Float16)zr[32 * k + off + j];
      a[j + 8] = (_Float16)zr[32 * k + 8 + off + j];
    }
    afrag[k] = a;
  }

  const int rbase = hi * 8;  // C/D: VGPR j <-> row rbase+j, col = lane%16
#pragma unroll
  for (int nt = 0; nt < 8; ++nt) {
    const int c = nt * 16 + mrow;
    // init accumulator with (1-beta)*Z so D = (1-beta)Z + beta*(Z@W)
    v8f acc;
#pragma unroll
    for (int j = 0; j < 8; ++j)
      acc[j] = (1.0f - kBeta) * zw[(rbase + j) * GD + c];
#pragma unroll
    for (int k = 0; k < 4; ++k) {
      // B fragment (32x16): lane holds column c; lane<16 K=32k+0..15,
      // lane>=16 K=32k+16..31 -> contiguous 32B LDS read
      const int k0 = 32 * k + hi * 16;
      v16h b = *(const v16h*)(wlds + c * GD + k0);
      acc = __builtin_amdgcn_wmma_f32_16x16x32_f16(
          /*neg_a=*/false, afrag[k], /*neg_b=*/false, b,
          /*c_mod=*/(short)0, acc, /*reuse_a=*/false, /*reuse_b=*/false);
    }
    // write D back into the (per-wave private) Z tile; safe: this tile's
    // C-init loads precede these stores and LDS is in-order per wave
#pragma unroll
    for (int j = 0; j < 8; ++j)
      zw[(rbase + j) * GD + c] = acc[j];
  }

  // coalesced writeback: 16 x global_store_b128 per lane, no divergence
  if (full) {
    for (int m = 0; m < 16; ++m)
      *(float4*)(out + (size_t)(r0 + m) * GD + c4) =
          *(const float4*)(zw + m * GD + c4);
  } else {
    for (int m = 0; m < 16; ++m)
      if (r0 + m < n)  // wave-uniform compare
        *(float4*)(out + (size_t)(r0 + m) * GD + c4) =
            *(const float4*)(zw + m * GD + c4);
  }
}

// ----------------------------------------------------------------------------
extern "C" void kernel_launch(void* const* d_in, const int* in_sizes, int n_in,
                              void* d_out, int out_size, void* d_ws, size_t ws_size,
                              hipStream_t stream) {
  const float*     x  = (const float*)d_in[0];
  const float*     x0 = (const float*)d_in[1];
  const float*     W  = (const float*)d_in[2];
  const long long* ei = (const long long*)d_in[3];  // int64 [2, E]
  float* out = (float*)d_out;

  const int n = in_sizes[0] / GD;
  const int e = in_sizes[3] / 2;
  const long long* src = ei;      // edge_index[0]
  const long long* dst = ei + e;  // edge_index[1]

  // workspace layout: deg[n] | dinv[n] | agg[n*GD]   (~52 MB)
  float* deg  = (float*)d_ws;
  float* dinv = deg + n;
  float* agg  = dinv + n;

  k_deg_init <<<(n + 255) / 256, 256, 0, stream>>>(deg, n);
  k_deg_count<<<(e + 255) / 256, 256, 0, stream>>>(dst, deg, e);
  k_dinv     <<<(n + 255) / 256, 256, 0, stream>>>(deg, dinv, n);

  long total = (long)n * GD;
  k_selfloop <<<(int)((total + 255) / 256), 256, 0, stream>>>(x, dinv, agg, total);

  // one wave32 per edge -> 8 edges per 256-thread block
  k_scatter  <<<(e + 7) / 8, 256, 0, stream>>>(x, dinv, src, dst, agg, e);

  // 128 rows per block
  k_out_wmma <<<(n + 127) / 128, 256, 0, stream>>>(agg, x0, W, out, n);
}
